// HPCrnn_67542655696962
// MI455X (gfx1250) — matrix operand order repaired and checked by reading, not compiled, not run on hardware
//
#include <hip/hip_runtime.h>
#include <hip/hip_bf16.h>
#include <stdint.h>
#include <stddef.h>

// ---------------------------------------------------------------------------
// Problem constants (from the reference)
// ---------------------------------------------------------------------------
#define ECN     1024
#define CA1N    1024
#define KDIM    1024
#define BSZ     8192
#define TRACK   100
#define CUELOC  10
#define NOISE_P 0.004f      // 0.04 * TS
#define EC5GAIN 1.0f        // 10.0 * TS

// LDS B-tile layout: 128 rows x (64 data bytes + 16 pad bytes)
#define BROW_BYTES  80
#define BTILE_ROWS  128
#define BTILE_BYTES (BTILE_ROWS * BROW_BYTES)   // 10240

typedef __attribute__((ext_vector_type(16))) __bf16 v16bf;
typedef __attribute__((ext_vector_type(8)))  __bf16 bf16x8;
typedef __attribute__((ext_vector_type(8)))  float  v8f;
typedef unsigned int u32x4 __attribute__((ext_vector_type(4)));
typedef int          i32x4 __attribute__((ext_vector_type(4)));
typedef int          i32x8 __attribute__((ext_vector_type(8)));

__device__ __forceinline__ float sigmoidf_fast(float x) {
    return 1.0f / (1.0f + __expf(-x));
}

__device__ __forceinline__ uint32_t hash32(uint32_t x) {
    x ^= x >> 16; x *= 0x7feb352dU;
    x ^= x >> 15; x *= 0x846ca68bU;
    x ^= x >> 16;
    return x;
}

// ---------------------------------------------------------------------------
// TDM: load one 128(rows) x 32(k) bf16 B tile from Wt (row stride 1024 elems)
// into LDS at lds_byte, padding each 64B row to 80B (bank-spread reads).
//
// D# group0: count=1 | lds_addr[63:32] | global_addr[120:64] | type=2[127:126]
// D# group1: data_size=1(2B) | pad_enable | pad_interval=3(16DW) |
//            pad_amount=3(4DW) | tensor_dim0=1024 | tensor_dim1=1024 |
//            tile_dim0=32 | tile_dim1=128 | tensor_dim0_stride=1024
// ---------------------------------------------------------------------------
__device__ __forceinline__ void tdm_load_b_tile(const __bf16* gptr, uint32_t lds_byte) {
    const uint64_t ga = (uint64_t)(uintptr_t)gptr;
    u32x4 g0;
    g0[0] = 1u;                                                // count = 1
    g0[1] = lds_byte;                                          // lds_addr
    g0[2] = (uint32_t)ga;                                      // global_addr[31:0]
    g0[3] = (uint32_t)((ga >> 32) & 0x01FFFFFFu) | (2u << 30); // addr[56:32] | type=2
    i32x8 g1 = { (int)0x06D10000,   // data_size=2B, pad_en, interval=16DW, amount=4DW
                 (int)0x04000000,   // tensor_dim0 = 1024 (bits 79:48, low half)
                 (int)0x04000000,   // tensor_dim0 hi=0 | tensor_dim1=1024 low
                 (int)0x00200000,   // tensor_dim1 hi=0 | tile_dim0 = 32
                 BTILE_ROWS,        // tile_dim1 = 128, tile_dim2 = 0
                 1024,              // tensor_dim0_stride = 1024
                 0, 0 };
    i32x4 z4 = {0, 0, 0, 0};
#if defined(__clang_major__) && (__clang_major__ >= 23)
    i32x8 z8 = {0, 0, 0, 0, 0, 0, 0, 0};
    __builtin_amdgcn_tensor_load_to_lds(g0, g1, z4, z4, z8, 0);
#else
    __builtin_amdgcn_tensor_load_to_lds(g0, g1, z4, z4, 0);
#endif
}

// ---------------------------------------------------------------------------
// WMMA fragment helpers (bf16 16x16x32, wave32 layouts per CDNA5 ISA)
// ---------------------------------------------------------------------------
__device__ __forceinline__ v16bf load_frag_a(const __bf16* __restrict__ rowptr,
                                             int k0, int half) {
    bf16x8 lo = *(const bf16x8*)(rowptr + k0 + half * 8);
    bf16x8 hi = *(const bf16x8*)(rowptr + k0 + 16 + half * 8);
    v16bf a;
#pragma unroll
    for (int i = 0; i < 8; ++i) { a[i] = lo[i]; a[8 + i] = hi[i]; }
    return a;
}

// ---------------------------------------------------------------------------
// Mainloop: block computes 128x128; 8 waves x (16 rows x 128 cols each).
// B tile (128x32) TDM-loaded into double-buffered LDS by wave 0 (TENSORcnt);
// one workgroup barrier per k-step:
//   wave0 wait(tensorcnt==0)  ->  barrier  ->  wave0 issue next TDM  -> compute
// A fragments are register-double-buffered global b128 loads.
// ---------------------------------------------------------------------------
__device__ __forceinline__ void wmma_16x128_tdm(const __bf16* __restrict__ arow,
                                                const __bf16* __restrict__ bbase, // wt + n0*KDIM
                                                int wid, int half, int l15,
                                                v8f acc[8]) {
    __shared__ __align__(128) unsigned char bsm[2 * BTILE_BYTES];
    const uint32_t lds0 = (uint32_t)(uintptr_t)(void*)&bsm[0];

    if (wid == 0) tdm_load_b_tile(bbase, lds0);            // prologue: buf 0
    v16bf a_cur = load_frag_a(arow, 0, half);
    int cur = 0;

    for (int k0 = 0; k0 < KDIM; k0 += 32) {
        if (wid == 0) __builtin_amdgcn_s_wait_tensorcnt(0); // buf(cur) landed
        __syncthreads();                                    // publishes buf(cur);
                                                            // proves buf(cur^1) reads done
        if (wid == 0 && k0 + 32 < KDIM)
            tdm_load_b_tile(bbase + (k0 + 32), lds0 + (cur ^ 1) * BTILE_BYTES);

        v16bf a_nxt = a_cur;
        if (k0 + 32 < KDIM) {
            __builtin_prefetch(arow + k0 + 64, 0, 3);       // global_prefetch_b8
            a_nxt = load_frag_a(arow, k0 + 32, half);
        }

        const unsigned char* bs = &bsm[cur * BTILE_BYTES];
#pragma unroll
        for (int nt = 0; nt < 8; ++nt) {
            const __bf16* bp =
                (const __bf16*)(bs + (nt * 16 + l15) * BROW_BYTES + half * 16);
            bf16x8 blo = *(const bf16x8*)bp;
            bf16x8 bhi = *(const bf16x8*)(bp + 16);
            v16bf b;
#pragma unroll
            for (int i = 0; i < 8; ++i) { b[i] = blo[i]; b[8 + i] = bhi[i]; }
            acc[nt] = __builtin_amdgcn_wmma_f32_16x16x32_bf16(
                false, a_cur, false, b, (short)0, acc[nt], false, false);
        }
        a_cur = a_nxt;
        cur ^= 1;
    }
}

// ---------------------------------------------------------------------------
// GEMM1: z = ec3_bf16 @ W1t;  ca1 = relu(drive*(1+sigmoid(z)) - bias)
// ---------------------------------------------------------------------------
__global__ __launch_bounds__(256)
void hpc_gemm1_ca1(const __bf16* __restrict__ ec3b,
                   const __bf16* __restrict__ w1t,
                   const float*  __restrict__ drive_t,
                   const float*  __restrict__ ca1bias,
                   __bf16*       __restrict__ ca1b) {
    const int lane = threadIdx.x & 31;
    const int wid  = threadIdx.x >> 5;
    const int m0   = (blockIdx.x >> 3) * 128 + wid * 16;   // 0..8176
    const int n0   = (blockIdx.x & 7) * 128;               // 0..896
    const int half = lane >> 4;
    const int l15  = lane & 15;

    v8f acc[8] = {v8f{}, v8f{}, v8f{}, v8f{}, v8f{}, v8f{}, v8f{}, v8f{}};
    wmma_16x128_tdm(ec3b + (size_t)(m0 + l15) * KDIM,
                    w1t + (size_t)n0 * KDIM, wid, half, l15, acc);

#pragma unroll
    for (int nt = 0; nt < 8; ++nt) {
        const int col  = n0 + nt * 16 + l15;
        const float d  = drive_t[col];
        const float bz = ca1bias[col];
#pragma unroll
        for (int r = 0; r < 8; ++r) {
            const int row = m0 + half * 8 + r;
            float v = d * (1.0f + sigmoidf_fast(acc[nt][r])) - bz;
            v = v > 0.0f ? v : 0.0f;
            ca1b[(size_t)row * CA1N + col] = (__bf16)v;
        }
    }
}

// ---------------------------------------------------------------------------
// GEMM2: u = ca1_bf16 @ W2t; fused ec5 squash + ec3 decay/cue/noise
// ---------------------------------------------------------------------------
__global__ __launch_bounds__(256)
void hpc_gemm2_update(const __bf16* __restrict__ ca1b,
                      const __bf16* __restrict__ w2t,
                      const float*  __restrict__ ec5bias,
                      const float*  __restrict__ cue,
                      float*        __restrict__ ec5,
                      float*        __restrict__ ec3f,
                      __bf16*       __restrict__ ec3b,
                      float cue_m, int t) {
    const int lane = threadIdx.x & 31;
    const int wid  = threadIdx.x >> 5;
    const int m0   = (blockIdx.x >> 3) * 128 + wid * 16;
    const int n0   = (blockIdx.x & 7) * 128;
    const int half = lane >> 4;
    const int l15  = lane & 15;

    v8f acc[8] = {v8f{}, v8f{}, v8f{}, v8f{}, v8f{}, v8f{}, v8f{}, v8f{}};
    wmma_16x128_tdm(ca1b + (size_t)(m0 + l15) * KDIM,
                    w2t + (size_t)n0 * KDIM, wid, half, l15, acc);

    const uint32_t tmix = (uint32_t)t * 0x9E3779B9u + 0x85EBCA6Bu;
#pragma unroll
    for (int nt = 0; nt < 8; ++nt) {
        const int col  = n0 + nt * 16 + l15;
        const float eb = ec5bias[col];
#pragma unroll
        for (int r = 0; r < 8; ++r) {
            const int row   = m0 + half * 8 + r;
            const size_t ix = (size_t)row * ECN + col;
            float e5 = ec5[ix] + EC5GAIN * acc[nt][r] + eb;
            e5 = 0.5f + 0.5f * sigmoidf_fast(4.0f * (e5 - 0.5f));
            float e3 = e5 * ec3f[ix] + cue_m * cue[ix];
            const uint32_t h = hash32((uint32_t)ix * 2654435761u + tmix);
            if ((float)(h >> 8) * (1.0f / 16777216.0f) < NOISE_P) e3 += 0.2f;
            ec5[ix]  = e5;
            ec3f[ix] = e3;
            ec3b[ix] = (__bf16)e3;
        }
    }
}

// ---------------------------------------------------------------------------
// drive[t][j] = sum_k exp(-(center_k - t)^2 / (2*sigma^2)) * wca3ca1[k][j]
// ---------------------------------------------------------------------------
__global__ void hpc_drive(const float* __restrict__ wca3ca1,
                          float* __restrict__ drive) {
    const int j = blockIdx.x * blockDim.x + threadIdx.x;
    const int t = blockIdx.y;
    const float step = (float)TRACK / (float)(KDIM - 1);
    float acc = 0.0f;
    for (int k = 0; k < KDIM; ++k) {
        const float c = (float)k * step - (float)t;
        acc += __expf(-c * c * 0.02f) * wca3ca1[(size_t)k * CA1N + j];
    }
    drive[(size_t)t * CA1N + j] = acc;
}

// wt[n][k] = bf16(w[k][n])  (transpose + fp32 -> bf16, one-time)
__global__ void hpc_convT(const float* __restrict__ w, __bf16* __restrict__ wt) {
    const int k = blockIdx.x * blockDim.x + threadIdx.x;
    const int n = blockIdx.y;
    wt[(size_t)n * KDIM + k] = (__bf16)w[(size_t)k * KDIM + n];
}

__global__ void hpc_init(const float* __restrict__ ec5_init,
                         float* __restrict__ ec5,
                         float* __restrict__ ec3f,
                         __bf16* __restrict__ ec3b) {
    const size_t i = (size_t)blockIdx.x * blockDim.x + threadIdx.x;
    ec5[i]  = ec5_init[i];
    ec3f[i] = 0.0f;
    ec3b[i] = (__bf16)0.0f;
}

// out[b][a] = ca1[b][:] . wca1act[:][a] + actbias[a]   (ACTNUM = 2)
__global__ void hpc_final(const __bf16* __restrict__ ca1b,
                          const float*  __restrict__ wact,
                          const float*  __restrict__ abias,
                          float* __restrict__ out) {
    const int id = blockIdx.x * blockDim.x + threadIdx.x;
    const int b  = id >> 1;
    const int a  = id & 1;
    const __bf16* row = ca1b + (size_t)b * CA1N;
    float acc = 0.0f;
    for (int j = 0; j < CA1N; ++j)
        acc += (float)row[j] * wact[(size_t)j * 2 + a];
    out[id] = acc + abias[a];
}

// ---------------------------------------------------------------------------
// Host-side orchestration
// ---------------------------------------------------------------------------
extern "C" void kernel_launch(void* const* d_in, const int* in_sizes, int n_in,
                              void* d_out, int out_size, void* d_ws, size_t ws_size,
                              hipStream_t stream) {
    (void)in_sizes; (void)n_in; (void)out_size; (void)ws_size;

    const float* cue      = (const float*)d_in[0];
    const float* ec5_init = (const float*)d_in[1];
    const float* wca3ca1  = (const float*)d_in[2];
    const float* wec3ca1  = (const float*)d_in[3];
    const float* wca1ec5  = (const float*)d_in[4];
    const float* wca1act  = (const float*)d_in[5];
    const float* ca1bias  = (const float*)d_in[6];
    const float* ec5bias  = (const float*)d_in[7];
    const float* actbias  = (const float*)d_in[8];
    float* out = (float*)d_out;

    unsigned char* ws = (unsigned char*)d_ws;
    size_t off = 0;
    float*  drive = (float*) (ws + off); off += (size_t)TRACK * CA1N * 4;
    __bf16* w1t   = (__bf16*)(ws + off); off += (size_t)KDIM * CA1N * 2;
    __bf16* w2t   = (__bf16*)(ws + off); off += (size_t)KDIM * ECN  * 2;
    float*  ec3f  = (float*) (ws + off); off += (size_t)BSZ * ECN * 4;
    float*  ec5   = (float*) (ws + off); off += (size_t)BSZ * ECN * 4;
    __bf16* ec3b  = (__bf16*)(ws + off); off += (size_t)BSZ * ECN * 2;
    __bf16* ca1b  = (__bf16*)(ws + off); off += (size_t)BSZ * CA1N * 2;

    hipLaunchKernelGGL(hpc_convT, dim3(KDIM / 256, KDIM), dim3(256), 0, stream,
                       wec3ca1, w1t);
    hipLaunchKernelGGL(hpc_convT, dim3(KDIM / 256, KDIM), dim3(256), 0, stream,
                       wca1ec5, w2t);
    hipLaunchKernelGGL(hpc_drive, dim3(CA1N / 256, TRACK), dim3(256), 0, stream,
                       wca3ca1, drive);
    hipLaunchKernelGGL(hpc_init, dim3((BSZ * ECN) / 256), dim3(256), 0, stream,
                       ec5_init, ec5, ec3f, ec3b);

    const dim3 gemm_grid((BSZ / 128) * (CA1N / 128));   // 64 * 8 = 512 blocks
    const dim3 gemm_blk(256);
    for (int t = 0; t < TRACK; ++t) {
        hipLaunchKernelGGL(hpc_gemm1_ca1, gemm_grid, gemm_blk, 0, stream,
                           ec3b, w1t, drive + (size_t)t * CA1N, ca1bias, ca1b);
        if (t + 1 < TRACK) {
            const float cue_m = (t == CUELOC) ? 1.0f : 0.0f;
            hipLaunchKernelGGL(hpc_gemm2_update, gemm_grid, gemm_blk, 0, stream,
                               ca1b, w2t, ec5bias, cue, ec5, ec3f, ec3b, cue_m, t);
        }
    }

    hipLaunchKernelGGL(hpc_final, dim3((BSZ * 2) / 256), dim3(256), 0, stream,
                       ca1b, wca1act, actbias, out);
}
